// WordSAGE_78847009620691
// MI455X (gfx1250) — matrix operand order repaired and checked by reading, not compiled, not run on hardware
//
#include <hip/hip_runtime.h>

// Problem constants (from reference)
#define NG 2500
#define NG_PAD 2528     // multiple of 32
#define NT 20000
#define M_PAD 20096     // 157 * 128  (block-tiled GEMM row padding)
#define NE 640000
#define SRC_DIM 128
#define DST_DIM 500
#define DST_PAD 512
#define HID 128
#define OUTD 128
#define NCLS 16

#define ASTRIDE 40      // LDS row stride in bf16 elems (32 data + 8 pad -> conflict-free b128)

typedef __attribute__((ext_vector_type(16))) __bf16 v16bf;
typedef __attribute__((ext_vector_type(8)))  __bf16 v8bf;
typedef __attribute__((ext_vector_type(8)))  float  v8f;

static __device__ inline v16bf bfcat(v8bf lo, v8bf hi) {
  return __builtin_shufflevector(lo, hi, 0, 1, 2, 3, 4, 5, 6, 7,
                                 8, 9, 10, 11, 12, 13, 14, 15);
}

// flat shared pointer -> group-segment-relative LDS byte offset (addr[31:0])
static __device__ inline unsigned lds_off(const void* p) {
  return (unsigned)(unsigned long long)p;
}

// async global->LDS 16B copy; INST_OFFSET applies to both sides (ISA 15.18.3 op 98)
static __device__ inline void async_b128(unsigned ldsAddr, const void* gaddr) {
  asm volatile("global_load_async_to_lds_b128 %0, %1, off"
               :: "v"(ldsAddr), "v"((unsigned long long)gaddr) : "memory");
}
static __device__ inline void async_b128_off16(unsigned ldsAddr, const void* gaddr) {
  asm volatile("global_load_async_to_lds_b128 %0, %1, off offset:16"
               :: "v"(ldsAddr), "v"((unsigned long long)gaddr) : "memory");
}

// ---------------------------------------------------------------- utilities
__global__ void zero_f4(float4* __restrict__ p, long n4) {
  long i = (long)blockIdx.x * blockDim.x + threadIdx.x;
  if (i < n4) p[i] = make_float4(0.f, 0.f, 0.f, 0.f);
}

// dense-count path: one packed-bf16 atomic per edge (exact small-int adds -> deterministic)
__global__ void edge_count(const int* __restrict__ esrc, const int* __restrict__ edst,
                           unsigned* __restrict__ C2, float* __restrict__ deg) {
  long e = (long)blockIdx.x * blockDim.x + threadIdx.x;
  if (e >= NE) return;
  int s = esrc[e];
  int d = edst[e];
  long idx = (long)d * NG_PAD + s;
  unsigned* addr = C2 + (idx >> 1);
  unsigned data = (idx & 1) ? 0x3F800000u : 0x00003F80u;  // bf16 1.0 in hi/lo half
  asm volatile("global_atomic_pk_add_bf16 %0, %1, off"
               :: "v"((unsigned long long)addr), "v"(data) : "memory");
  atomicAdd(deg + d, 1.0f);
}

__global__ void inv_deg(const float* __restrict__ deg, float* __restrict__ inv) {
  int i = blockIdx.x * blockDim.x + threadIdx.x;
  if (i < M_PAD) inv[i] = 1.0f / fmaxf(deg[i], 1.0f);
}

// ---- fallback path (small ws): per-edge feature scatter
__global__ void edge_scatter(const float* __restrict__ gene,
                             const int* __restrict__ esrc,
                             const int* __restrict__ edst,
                             float* __restrict__ agg,
                             float* __restrict__ cnt) {
  long idx = (long)blockIdx.x * blockDim.x + threadIdx.x;
  long e = idx >> 5;
  int  c = (int)(idx & 31);
  if (e >= NE) return;
  int s = esrc[e];
  int d = edst[e];
  float4 g = ((const float4*)(gene + (long)s * SRC_DIM))[c];
  float* a = agg + (long)d * SRC_DIM + c * 4;
  atomicAdd(a + 0, g.x);
  atomicAdd(a + 1, g.y);
  atomicAdd(a + 2, g.z);
  atomicAdd(a + 3, g.w);
  if (c == 0) atomicAdd(cnt + d, 1.0f);
}

__global__ void norm_agg(const float* __restrict__ agg, const float* __restrict__ cnt,
                         __bf16* __restrict__ aggBF) {
  long idx = (long)blockIdx.x * blockDim.x + threadIdx.x;
  if (idx >= (long)M_PAD * SRC_DIM) return;
  long i = idx >> 7;
  float v = (i < NT) ? agg[idx] / fmaxf(cnt[i], 1.0f) : 0.0f;
  aggBF[idx] = (__bf16)v;
}

// train_feats f32 [NT x 500] -> bf16 [M_PAD x 512] zero-padded both dims
__global__ void conv_train(const float* __restrict__ T, __bf16* __restrict__ TB) {
  long idx = (long)blockIdx.x * blockDim.x + threadIdx.x;
  if (idx >= (long)M_PAD * DST_PAD) return;
  long i = idx / DST_PAD;
  int  k = (int)(idx - i * DST_PAD);
  float v = (i < NT && k < DST_DIM) ? T[i * DST_DIM + k] : 0.0f;
  TB[idx] = (__bf16)v;
}

// W f32 [K x N] -> WT bf16 [N x Kpad] (transposed, zero-padded K)
__global__ void conv_weight_T(const float* __restrict__ W, __bf16* __restrict__ WT,
                              int K, int Kpad, int N_) {
  int idx = blockIdx.x * blockDim.x + threadIdx.x;
  if (idx >= N_ * Kpad) return;
  int n = idx / Kpad;
  int k = idx - n * Kpad;
  float v = (k < K) ? W[(long)k * N_ + n] : 0.0f;
  WT[idx] = (__bf16)v;
}

// ------------------------------------------------- LDS-tiled WMMA GEMM (N=128)
// D[M_PAD x 128] = act( scale(row) * (A1 @ B1T^T + A2 @ B2T^T) + bias )
// Block: 128x128 tile, 8 waves, each wave 32(M) x 64(N) via 8 accumulators.
// A/B K-slices (width 32) double-buffered in LDS via async global->LDS copies.
template <int K1, int K2, bool RELU, bool HAS_BIAS, bool HAS_SCALE>
__global__ __launch_bounds__(256) void gemm_tiled(
    const __bf16* __restrict__ A1, const __bf16* __restrict__ B1T,
    const __bf16* __restrict__ A2, const __bf16* __restrict__ B2T,
    const float* __restrict__ bias, const float* __restrict__ rscale,
    __bf16* __restrict__ Dbf) {
  __shared__ __bf16 As[2][128 * ASTRIDE];
  __shared__ __bf16 Bs[2][128 * ASTRIDE];

  const int tid  = threadIdx.x;
  const int lane = tid & 31;
  const int wid  = tid >> 5;
  const int mw   = wid & 3;    // 32-row group within block
  const int nw   = wid >> 2;   // 64-col half
  const int half = lane >> 4;
  const int mn   = lane & 15;
  const long rowBlk = (long)blockIdx.x * 128;

  const int srow = tid >> 1;   // staging row/col (0..127)
  const int sseg = tid & 1;    // 16-element segment

  const unsigned stageByte = (unsigned)(srow * ASTRIDE + sseg * 16) * 2u;
  unsigned ldsA[2], ldsB[2];
  ldsA[0] = lds_off(&As[0][0]) + stageByte;
  ldsA[1] = lds_off(&As[1][0]) + stageByte;
  ldsB[0] = lds_off(&Bs[0][0]) + stageByte;
  ldsB[1] = lds_off(&Bs[1][0]) + stageByte;

  v8f acc[2][4];
#pragma unroll
  for (int s = 0; s < 2; ++s)
#pragma unroll
    for (int j = 0; j < 4; ++j) acc[s][j] = (v8f){};

  constexpr int NK1 = K1 / 32;
  constexpr int NK  = K1 / 32 + K2 / 32;

  // issue one K-slice (4 async b128 per thread; ASYNCcnt += 4 per wave)
  auto issue_stage = [&](int kc, int buf) {
    const __bf16* Ap;
    const __bf16* Bp;
    long K, kco;
    if (kc < NK1) {
      Ap = A1; Bp = B1T; K = K1; kco = (long)kc * 32;
    } else {
      Ap = A2; Bp = B2T; K = K2; kco = (long)(kc - NK1) * 32;
    }
    const __bf16* asrc = Ap + (rowBlk + srow) * K + kco + sseg * 16;
    const __bf16* bsrc = Bp + (long)srow * K + kco + sseg * 16;
    async_b128(ldsA[buf], asrc);
    async_b128_off16(ldsA[buf], asrc);
    async_b128(ldsB[buf], bsrc);
    async_b128_off16(ldsB[buf], bsrc);
  };

  issue_stage(0, 0);
  for (int kc = 0; kc < NK; ++kc) {
    const int cur = kc & 1;
    if (kc + 1 < NK) {
      issue_stage(kc + 1, cur ^ 1);  // prefetch next slice into other buffer
      asm volatile("s_wait_asynccnt 4" ::: "memory");  // retire current slice only
    } else {
      asm volatile("s_wait_asynccnt 0" ::: "memory");
    }
    __syncthreads();  // all waves' slices visible
#pragma unroll
    for (int s = 0; s < 2; ++s) {
      const __bf16* ap = &As[cur][(mw * 32 + s * 16 + mn) * ASTRIDE + half * 8];
      v16bf a = bfcat(*(const v8bf*)ap, *(const v8bf*)(ap + 16));
#pragma unroll
      for (int j = 0; j < 4; ++j) {
        const __bf16* bp = &Bs[cur][(nw * 64 + j * 16 + mn) * ASTRIDE + half * 8];
        v16bf b = bfcat(*(const v8bf*)bp, *(const v8bf*)(bp + 16));
        acc[s][j] = __builtin_amdgcn_wmma_f32_16x16x32_bf16(
            false, a, false, b, (short)0, acc[s][j], false, false);
      }
    }
    __syncthreads();  // finished reading 'cur' before it is refilled
  }

  // ---- epilogue: optional row-scale, bias, relu; bf16 store
#pragma unroll
  for (int s = 0; s < 2; ++s) {
    float rs[8];
#pragma unroll
    for (int r = 0; r < 8; ++r) {
      const long row = rowBlk + mw * 32 + s * 16 + half * 8 + r;
      rs[r] = HAS_SCALE ? rscale[row] : 1.0f;
    }
#pragma unroll
    for (int j = 0; j < 4; ++j) {
      const int col = nw * 64 + j * 16 + mn;
      const float bv = HAS_BIAS ? bias[col] : 0.0f;
#pragma unroll
      for (int r = 0; r < 8; ++r) {
        const long row = rowBlk + mw * 32 + s * 16 + half * 8 + r;
        float v = acc[s][j][r] * rs[r] + bv;
        if (RELU) v = fmaxf(v, 0.0f);
        Dbf[row * 128 + col] = (__bf16)v;
      }
    }
  }
}

// ---------------------------------------- final layer: N=16, f32 out to d_out
__global__ __launch_bounds__(32) void gemm_out(
    const __bf16* __restrict__ A, const __bf16* __restrict__ BT,
    const float* __restrict__ bias, float* __restrict__ D) {
  const int lane = threadIdx.x & 31;
  const int rowBase = blockIdx.x * 16;
  const int half = lane >> 4;
  const int mn = lane & 15;

  v8f acc = {};
  const __bf16* pa = A + (long)(rowBase + mn) * OUTD + half * 8;
  const __bf16* pb = BT + (long)mn * OUTD + half * 8;
#pragma unroll
  for (int kc = 0; kc < OUTD / 32; ++kc) {
    v16bf a = bfcat(*(const v8bf*)(pa + kc * 32), *(const v8bf*)(pa + kc * 32 + 16));
    v16bf b = bfcat(*(const v8bf*)(pb + kc * 32), *(const v8bf*)(pb + kc * 32 + 16));
    acc = __builtin_amdgcn_wmma_f32_16x16x32_bf16(
        false, a, false, b, (short)0, acc, false, false);
  }
  const float bv = bias[mn];
#pragma unroll
  for (int r = 0; r < 8; ++r) {
    const long row = rowBase + half * 8 + r;
    D[row * NCLS + mn] = acc[r] + bv;
  }
}

// ---------------------------------------------------------------- launcher
extern "C" void kernel_launch(void* const* d_in, const int* in_sizes, int n_in,
                              void* d_out, int out_size, void* d_ws, size_t ws_size,
                              hipStream_t stream) {
  const float* gene  = (const float*)d_in[0];
  const float* train = (const float*)d_in[1];
  const int*   esrc  = (const int*)d_in[2];
  const int*   edst  = (const int*)d_in[3];
  const float* W1n   = (const float*)d_in[4];
  const float* W1s   = (const float*)d_in[5];
  const float* b1    = (const float*)d_in[6];
  const float* W2n   = (const float*)d_in[7];
  const float* W2s   = (const float*)d_in[8];
  const float* b2    = (const float*)d_in[9];
  const float* Wc1   = (const float*)d_in[10];
  const float* bc1   = (const float*)d_in[11];
  const float* Wc2   = (const float*)d_in[12];
  const float* bc2   = (const float*)d_in[13];

  char* ws = (char*)d_ws;
  size_t off = 0;
  auto alloc = [&](size_t bytes) -> void* {
    void* p = ws + off;
    off += bytes;
    off = (off + 255) & ~(size_t)255;
    return p;
  };

  float*  deg     = (float*)alloc((size_t)M_PAD * 4);
  float*  invDeg  = (float*)alloc((size_t)M_PAD * 4);
  __bf16* trainBF = (__bf16*)alloc((size_t)M_PAD * DST_PAD * 2);
  __bf16* aggBF   = (__bf16*)alloc((size_t)M_PAD * SRC_DIM * 2);
  __bf16* h1      = (__bf16*)alloc((size_t)M_PAD * HID * 2);
  __bf16* h2      = (__bf16*)alloc((size_t)M_PAD * OUTD * 2);
  __bf16* h3      = (__bf16*)alloc((size_t)M_PAD * OUTD * 2);
  __bf16* W1sT    = (__bf16*)alloc((size_t)HID * DST_PAD * 2);
  __bf16* W1nT    = (__bf16*)alloc((size_t)HID * SRC_DIM * 2);
  __bf16* W2sT    = (__bf16*)alloc((size_t)OUTD * HID * 2);
  __bf16* W2nT    = (__bf16*)alloc((size_t)OUTD * SRC_DIM * 2);
  __bf16* Wc1T    = (__bf16*)alloc((size_t)OUTD * OUTD * 2);
  __bf16* Wc2T    = (__bf16*)alloc((size_t)NCLS * OUTD * 2);
  __bf16* geneT   = (__bf16*)alloc((size_t)SRC_DIM * NG_PAD * 2);
  float*  aggF    = (float*)alloc((size_t)NT * SRC_DIM * 4);   // fallback only
  float*  cnt     = (float*)alloc((size_t)NT * 4);             // fallback only
  const size_t C_bytes = (size_t)M_PAD * NG_PAD * 2;           // bf16 count matrix
  const bool dense_ok = (off + C_bytes) <= ws_size;
  __bf16* Cbf = (__bf16*)alloc(C_bytes);

  const int TPB = 256;
  auto blocks = [](long n, int tpb) { return (unsigned)((n + tpb - 1) / tpb); };

  // ---------------- aggregation -> aggBF [M_PAD x 128] bf16
  if (dense_ok) {
    // agg = diag(1/deg) * C @ gene  with C built by 640k packed-bf16 atomics
    zero_f4<<<blocks((long)(C_bytes / 16), TPB), TPB, 0, stream>>>(
        (float4*)Cbf, (long)(C_bytes / 16));
    zero_f4<<<blocks(M_PAD / 4, TPB), TPB, 0, stream>>>((float4*)deg, M_PAD / 4);
    edge_count<<<blocks(NE, TPB), TPB, 0, stream>>>(esrc, edst, (unsigned*)Cbf, deg);
    inv_deg<<<blocks(M_PAD, TPB), TPB, 0, stream>>>(deg, invDeg);
    conv_weight_T<<<blocks(SRC_DIM * NG_PAD, TPB), TPB, 0, stream>>>(
        gene, geneT, NG, NG_PAD, SRC_DIM);
    gemm_tiled<NG_PAD, 0, false, false, true><<<M_PAD / 128, 256, 0, stream>>>(
        Cbf, geneT, nullptr, nullptr, nullptr, invDeg, aggBF);
  } else {
    // fallback: per-edge feature scatter with f32 atomics
    zero_f4<<<blocks((long)NT * SRC_DIM / 4, TPB), TPB, 0, stream>>>(
        (float4*)aggF, (long)NT * SRC_DIM / 4);
    zero_f4<<<blocks(NT / 4, TPB), TPB, 0, stream>>>((float4*)cnt, NT / 4);
    edge_scatter<<<blocks((long)NE * 32, TPB), TPB, 0, stream>>>(gene, esrc, edst, aggF, cnt);
    norm_agg<<<blocks((long)M_PAD * SRC_DIM, TPB), TPB, 0, stream>>>(aggF, cnt, aggBF);
  }

  // ---------------- precision/layout conversion
  conv_train<<<blocks((long)M_PAD * DST_PAD, TPB), TPB, 0, stream>>>(train, trainBF);
  conv_weight_T<<<blocks(HID * DST_PAD, TPB), TPB, 0, stream>>>(W1s, W1sT, DST_DIM, DST_PAD, HID);
  conv_weight_T<<<blocks(HID * SRC_DIM, TPB), TPB, 0, stream>>>(W1n, W1nT, SRC_DIM, SRC_DIM, HID);
  conv_weight_T<<<blocks(OUTD * HID, TPB), TPB, 0, stream>>>(W2s, W2sT, HID, HID, OUTD);
  conv_weight_T<<<blocks(OUTD * SRC_DIM, TPB), TPB, 0, stream>>>(W2n, W2nT, SRC_DIM, SRC_DIM, OUTD);
  conv_weight_T<<<blocks(OUTD * OUTD, TPB), TPB, 0, stream>>>(Wc1, Wc1T, OUTD, OUTD, OUTD);
  conv_weight_T<<<blocks(NCLS * OUTD, TPB), TPB, 0, stream>>>(Wc2, Wc2T, OUTD, OUTD, NCLS);

  // ---------------- WMMA layer chain
  gemm_tiled<DST_PAD, SRC_DIM, true, true, false><<<M_PAD / 128, 256, 0, stream>>>(
      trainBF, W1sT, aggBF, W1nT, b1, nullptr, h1);
  gemm_tiled<HID, SRC_DIM, true, true, false><<<M_PAD / 128, 256, 0, stream>>>(
      h1, W2sT, aggBF, W2nT, b2, nullptr, h2);
  gemm_tiled<OUTD, 0, true, true, false><<<M_PAD / 128, 256, 0, stream>>>(
      h2, Wc1T, nullptr, nullptr, bc1, nullptr, h3);
  gemm_out<<<NT / 16, 32, 0, stream>>>(h3, Wc2T, bc2, (float*)d_out);
}
